// Upsample_18279380811797
// MI455X (gfx1250) — compile-verified, matrix operand order checked by reading, not run on hardware
//
#include <hip/hip_runtime.h>

typedef __attribute__((ext_vector_type(16))) __bf16 v16bf;
typedef __attribute__((ext_vector_type(8)))  __bf16 v8bf;
typedef __attribute__((ext_vector_type(4)))  __bf16 v4bf;
typedef __attribute__((ext_vector_type(8)))  float  v8f;

#define N_ROWS 65536
#define CIN    256
#define COUT   128
#define KOFF   8
#define TOTAL_ROWS (KOFF * N_ROWS)   // 524288

__device__ __forceinline__ v8bf ld8(const __bf16* p) {
  return *reinterpret_cast<const v8bf*>(p);
}

__device__ __forceinline__ v16bf pack16(v8bf lo, v8bf hi) {
  v16bf r;
#pragma unroll
  for (int i = 0; i < 8; ++i) { r[i] = lo[i]; r[8 + i] = hi[i]; }
  return r;
}

// ---- fp32 -> bf16 convert of x (vectorized: float4 in, 8B out) ----
__global__ __launch_bounds__(256) void k_convert_x(const float* __restrict__ x,
                                                   __bf16* __restrict__ Xb) {
  int i = (blockIdx.x * 256 + threadIdx.x) * 4;
  float4 v = *reinterpret_cast<const float4*>(x + i);
  v4bf o;
  o[0] = (__bf16)v.x; o[1] = (__bf16)v.y; o[2] = (__bf16)v.z; o[3] = (__bf16)v.w;
  *reinterpret_cast<v4bf*>(Xb + i) = o;
}

// ---- W[k][c][d] fp32 -> WbT[k][d][c] bf16 (transpose so B K-runs are contiguous) ----
__global__ __launch_bounds__(256) void k_convert_w(const float* __restrict__ W,
                                                   __bf16* __restrict__ WbT) {
  int i = blockIdx.x * 256 + threadIdx.x;     // i = k*32768 + d*256 + c
  int k = i >> 15, r = i & 32767, d = r >> 8, c = r & 255;
  WbT[i] = (__bf16)W[k * (CIN * COUT) + c * COUT + d];
}

// ---- bf16 WMMA GEMM: wave computes a 16x128 tile; fused channel-stat partials ----
__global__ __launch_bounds__(256) void k_gemm(const __bf16* __restrict__ Xb,
                                              const __bf16* __restrict__ WbT,
                                              float* __restrict__ outRaw,
                                              float* __restrict__ partials) {
  __shared__ float red[2 * COUT];
  red[threadIdx.x] = 0.0f;
  __syncthreads();

  const int lane  = threadIdx.x & 31;
  const int wave  = threadIdx.x >> 5;           // 0..7  (8 waves per block)
  const int k     = blockIdx.y;                 // 0..7  (kernel offset)
  const int mtile = blockIdx.x * 8 + wave;      // 0..4095
  const int mbase = mtile * 16;
  const int ln = lane & 15;
  const int hi = lane >> 4;

  const __bf16* Arow = Xb + (mbase + ln) * CIN + hi * 8;
  const __bf16* Bk   = WbT + k * (COUT * CIN);

  v8f acc[8] = {};

  for (int kb = 0; kb < CIN; kb += 32) {
    v16bf a = pack16(ld8(Arow + kb), ld8(Arow + kb + 16));
    if (kb + 32 < CIN) __builtin_prefetch(Arow + kb + 32, 0, 0);
#pragma unroll
    for (int t = 0; t < 8; ++t) {
      const __bf16* Brow = Bk + (t * 16 + ln) * CIN + hi * 8;
      v16bf b = pack16(ld8(Brow + kb), ld8(Brow + kb + 16));
      acc[t] = __builtin_amdgcn_wmma_f32_16x16x32_bf16(
          false, a, false, b, (short)0, acc[t], false, false);
    }
  }

  // store raw GEMM output + accumulate per-channel sum / sum-of-squares
  const long rowbase = (long)k * N_ROWS + mbase;
#pragma unroll
  for (int t = 0; t < 8; ++t) {
    const int col = t * 16 + ln;
    float s = 0.0f, s2 = 0.0f;
#pragma unroll
    for (int v = 0; v < 8; ++v) {
      float val = acc[t][v];
      s  += val;
      s2 += val * val;
      outRaw[(rowbase + v + 8 * hi) * COUT + col] = val;
    }
    atomicAdd(&red[col], s);          // ds_add_f32, block-local
    atomicAdd(&red[COUT + col], s2);
  }
  __syncthreads();
  const int bid = blockIdx.y * gridDim.x + blockIdx.x;
  partials[(size_t)bid * (2 * COUT) + threadIdx.x] = red[threadIdx.x];
}

// ---- reduce partials -> per-channel scale/shift (folds mean/var/gamma/beta) ----
__global__ __launch_bounds__(256) void k_finalize(const float* __restrict__ partials,
                                                  const float* __restrict__ gamma,
                                                  const float* __restrict__ beta,
                                                  float* __restrict__ ss,
                                                  int nblocks) {
  __shared__ float tot[2 * COUT];
  int j = threadIdx.x;
  float s = 0.0f;
  for (int b = 0; b < nblocks; ++b) s += partials[(size_t)b * (2 * COUT) + j];
  tot[j] = s;
  __syncthreads();
  if (j < COUT) {
    const float inv = 1.0f / (float)TOTAL_ROWS;
    float mean = tot[j] * inv;
    float var  = tot[COUT + j] * inv - mean * mean;
    float sc   = gamma[j] * rsqrtf(var + 1e-5f);
    ss[j]        = sc;
    ss[COUT + j] = beta[j] - mean * sc;
  }
}

// ---- in-place normalize + ReLU, float4 vectorized ----
__global__ __launch_bounds__(256) void k_normalize(float* __restrict__ out,
                                                   const float* __restrict__ ss) {
  long i = (long)blockIdx.x * 256 + threadIdx.x;
  float4 v = reinterpret_cast<float4*>(out)[i];
  int c = (int)((i * 4) & (COUT - 1));
  v.x = fmaxf(fmaf(v.x, ss[c],     ss[COUT + c]),     0.0f);
  v.y = fmaxf(fmaf(v.y, ss[c + 1], ss[COUT + c + 1]), 0.0f);
  v.z = fmaxf(fmaf(v.z, ss[c + 2], ss[COUT + c + 2]), 0.0f);
  v.w = fmaxf(fmaf(v.w, ss[c + 3], ss[COUT + c + 3]), 0.0f);
  reinterpret_cast<float4*>(out)[i] = v;
}

extern "C" void kernel_launch(void* const* d_in, const int* in_sizes, int n_in,
                              void* d_out, int out_size, void* d_ws, size_t ws_size,
                              hipStream_t stream) {
  const float* x     = (const float*)d_in[0];
  const float* W     = (const float*)d_in[1];
  const float* gamma = (const float*)d_in[2];
  const float* beta  = (const float*)d_in[3];
  float* out = (float*)d_out;

  // workspace layout: Xb (32 MB) | WbT (0.5 MB) | partials (4 MB) | scale/shift (1 KB)
  __bf16* Xb  = (__bf16*)d_ws;
  __bf16* WbT = Xb + (size_t)N_ROWS * CIN;
  float*  partials = (float*)(WbT + (size_t)KOFF * COUT * CIN);
  const int gemm_blocks = (N_ROWS / 128) * KOFF;      // 512 * 8 = 4096
  float*  ss = partials + (size_t)gemm_blocks * (2 * COUT);

  k_convert_x<<<dim3(N_ROWS * CIN / (256 * 4)), dim3(256), 0, stream>>>(x, Xb);
  k_convert_w<<<dim3(KOFF * COUT * CIN / 256), dim3(256), 0, stream>>>(W, WbT);
  k_gemm<<<dim3(N_ROWS / 128, KOFF), dim3(256), 0, stream>>>(Xb, WbT, out, partials);
  k_finalize<<<dim3(1), dim3(256), 0, stream>>>(partials, gamma, beta, ss, gemm_blocks);
  k_normalize<<<dim3(TOTAL_ROWS * COUT / (256 * 4)), dim3(256), 0, stream>>>(out, ss);

  (void)in_sizes; (void)n_in; (void)out_size; (void)ws_size;
}
